// GraphDistillOperator_28613072126556
// MI455X (gfx1250) — matrix operand order repaired and checked by reading, not compiled, not run on hardware
//
#include <hip/hip_runtime.h>
#include <hip/hip_bf16.h>
#include <math.h>

#define GN 16384   // nodes
#define GD 128     // feature dim
#define GH 256     // hidden dim

typedef __attribute__((ext_vector_type(8)))  float  v8f;
typedef __attribute__((ext_vector_type(16))) __bf16 v16bf;
typedef __attribute__((ext_vector_type(4)))  int    v4i;

union BFrag {
  v16bf v;
  unsigned int u[8];
  uint4 q[2];
};

__device__ __forceinline__ unsigned int f2bf(float f) {
  unsigned int x = __float_as_uint(f);
  return (x + 0x7FFFu + ((x >> 16) & 1u)) >> 16;   // RNE f32 -> bf16 bits
}
__device__ __forceinline__ float bf2f(unsigned int h) {
  return __uint_as_float(h << 16);
}
__device__ __forceinline__ unsigned int pack01(int a, int b) {
  // a,b in {0,1}: bf16(a) | bf16(b)<<16
  return (unsigned)a * 0x3F80u + (unsigned)b * 0x3F800000u;
}
__device__ __forceinline__ v8f v8fzero() {
  v8f z;
#pragma unroll
  for (int i = 0; i < 8; ++i) z[i] = 0.0f;
  return z;
}
__device__ __forceinline__ v8f wmma_bf16(const BFrag& a, const BFrag& b, v8f c) {
  return __builtin_amdgcn_wmma_f32_16x16x32_bf16(false, a.v, false, b.v,
                                                 (short)0, c, false, false);
}
// split 16 floats (ascending K order) into packed bf16 hi/lo fragments
__device__ __forceinline__ void splitpack(const float* x, BFrag& hi, BFrag& lo) {
#pragma unroll
  for (int v = 0; v < 8; ++v) {
    unsigned int h0 = f2bf(x[2 * v]);
    unsigned int h1 = f2bf(x[2 * v + 1]);
    unsigned int l0 = f2bf(x[2 * v] - bf2f(h0));
    unsigned int l1 = f2bf(x[2 * v + 1] - bf2f(h1));
    hi.u[v] = h0 | (h1 << 16);
    lo.u[v] = l0 | (l1 << 16);
  }
}
// load 16 floats with A-fragment gap pattern: p[0..7], p[16..23]
__device__ __forceinline__ void load16gap(const float* p, float* o) {
  const float4* q = (const float4*)p;
  float4 a = q[0], b = q[1], c = q[4], d = q[5];
  o[0]=a.x; o[1]=a.y; o[2]=a.z; o[3]=a.w; o[4]=b.x; o[5]=b.y; o[6]=b.z; o[7]=b.w;
  o[8]=c.x; o[9]=c.y; o[10]=c.z; o[11]=c.w; o[12]=d.x; o[13]=d.y; o[14]=d.z; o[15]=d.w;
}

// CDNA5 async global->LDS copy (16B per lane), tracked by ASYNCcnt.
__device__ __forceinline__ void copy16_async(unsigned lds_byte_addr,
                                             const void* gsrc) {
  asm volatile("global_load_async_to_lds_b128 %0, %1, off"
               :: "v"(lds_byte_addr), "v"(gsrc)
               : "memory");
}
__device__ __forceinline__ void wait_async() {
  asm volatile("s_wait_asynccnt 0x0" ::: "memory");
}

// ---------------------------------------------------------------------------
// Kernel 0a: features[N][D] f32 -> transposed bf16 hi/lo planes featT[D][N]
// ---------------------------------------------------------------------------
__global__ void prep_kernel(const float* __restrict__ feat,
                            unsigned short* __restrict__ fTh,
                            unsigned short* __restrict__ fTl) {
  size_t idx = (size_t)blockIdx.x * 256 + threadIdx.x;   // N*D elements
  int i = (int)(idx >> 7);
  int d = (int)(idx & 127);
  float f = feat[idx];
  unsigned int h = f2bf(f);
  unsigned int l = f2bf(f - bf2f(h));
  size_t o = (size_t)d * GN + i;
  fTh[o] = (unsigned short)h;
  fTl[o] = (unsigned short)l;
}

// ---------------------------------------------------------------------------
// Kernel 0b: combined W'' = [Wd rows | Wo rows] -> bf16 hi/lo planes [H][384]
// ---------------------------------------------------------------------------
__global__ void prepw_kernel(const float* __restrict__ Wd,
                             const float* __restrict__ Wo,
                             unsigned short* __restrict__ Wbh,
                             unsigned short* __restrict__ Wbl) {
  int idx = blockIdx.x * 256 + threadIdx.x;   // 256*384 elements
  int h = idx / 384;
  int k = idx - h * 384;
  float v = (k < 256) ? Wd[(size_t)h * 256 + k] : Wo[(size_t)h * GD + (k - 256)];
  unsigned int hi = f2bf(v);
  Wbh[idx] = (unsigned short)hi;
  Wbl[idx] = (unsigned short)f2bf(v - bf2f(hi));
}

// ---------------------------------------------------------------------------
// Kernel 1: tail = adj @ features (bf16 hi/lo split), deg = adj @ ones.
// Grid: 256 blocks = 128 row-blocks x 2 K-splits. Block: 128 thr = 4 waves.
// Wave covers 32 rows (two 16-row M tiles), K range 8192 per split.
// B slab staged by async-LDS copies, double buffered, 1 barrier / k-step.
// adj loads non-temporal (1 GiB stream must not evict L2-resident featT/W).
// ---------------------------------------------------------------------------
__global__ __launch_bounds__(128, 1)
void gemm_adj_kernel(const int* __restrict__ adj,
                     const unsigned short* __restrict__ fTh,
                     const unsigned short* __restrict__ fTl,
                     float* __restrict__ tail0,   // [2][N][D]
                     float* __restrict__ deg0) {  // [2][N]
  __shared__ uint4 lds[2048];   // 32 KB: two 16 KB buffers (hi 8K | lo 8K)
  const int tid    = threadIdx.x;
  const int lane   = tid & 31;
  const int lrow   = lane & 15;
  const int rowblk = blockIdx.x & 127;
  const int ksplit = blockIdx.x >> 7;
  const int kbase  = ksplit * 8192;
  const int rowBase = rowblk * 128 + (tid >> 5) * 32;
  const int khalfA  = (lane < 16) ? 0 : 8;    // int offset into adj row
  const int ldsHalf = (lane < 16) ? 0 : 1;    // uint4 offset within LDS row
  const unsigned ldsbase = (unsigned)(uintptr_t)&lds[0];

  BFrag bones;                                // B tile: column 0 == 1.0
  {
    unsigned int bv = (lrow == 0) ? 0x3F803F80u : 0u;
#pragma unroll
    for (int j = 0; j < 8; ++j) bones.u[j] = bv;
  }

  v8f c[2][8];
  v8f cdeg[2];
#pragma unroll
  for (int rt = 0; rt < 2; ++rt) {
    cdeg[rt] = v8fzero();
#pragma unroll
    for (int t = 0; t < 8; ++t) c[rt][t] = v8fzero();
  }

  // stage slab for chunk k0 into LDS buffer `buf` (thread t = featT dim row t)
  auto stage = [&](int buf, int k0) {
    unsigned dh = ldsbase + (unsigned)buf * 16384u + (unsigned)tid * 64u;
    const uint4* gh = (const uint4*)(fTh + (size_t)tid * GN + k0);
    const uint4* gl = (const uint4*)(fTl + (size_t)tid * GN + k0);
#pragma unroll
    for (int j = 0; j < 4; ++j) {
      copy16_async(dh + j * 16u, gh + j);
      copy16_async(dh + 8192u + j * 16u, gl + j);
    }
  };

  stage(0, kbase);
  wait_async();
  __syncthreads();

  for (int i = 0; i < 256; ++i) {
    const int k0  = kbase + i * 32;
    const int cur = i & 1;
    if (i + 1 < 256) stage(cur ^ 1, k0 + 32);   // overlap with compute

    // A fragments: adj rows, 0/1 -> bf16, cacheline-coalesced, non-temporal
    BFrag a[2];
#pragma unroll
    for (int rt = 0; rt < 2; ++rt) {
      const v4i* ap =
          (const v4i*)(adj + (size_t)(rowBase + rt * 16 + lrow) * GN + k0 + khalfA);
      v4i i0 = __builtin_nontemporal_load(ap);
      v4i i1 = __builtin_nontemporal_load(ap + 1);
      v4i i2 = __builtin_nontemporal_load(ap + 4);
      v4i i3 = __builtin_nontemporal_load(ap + 5);
      __builtin_prefetch((const void*)(ap + 16));   // 2 k-chunks ahead
      a[rt].u[0] = pack01(i0.x, i0.y);
      a[rt].u[1] = pack01(i0.z, i0.w);
      a[rt].u[2] = pack01(i1.x, i1.y);
      a[rt].u[3] = pack01(i1.z, i1.w);
      a[rt].u[4] = pack01(i2.x, i2.y);
      a[rt].u[5] = pack01(i2.z, i2.w);
      a[rt].u[6] = pack01(i3.x, i3.y);
      a[rt].u[7] = pack01(i3.z, i3.w);
    }

    cdeg[0] = wmma_bf16(a[0], bones, cdeg[0]);
    cdeg[1] = wmma_bf16(a[1], bones, cdeg[1]);

    const int bufBase = cur * 1024;
#pragma unroll
    for (int tl = 0; tl < 8; ++tl) {
      BFrag bh, bl;
      int base = bufBase + (tl * 16 + lrow) * 4 + ldsHalf;
      bh.q[0] = lds[base];       bh.q[1] = lds[base + 2];
      bl.q[0] = lds[512 + base]; bl.q[1] = lds[512 + base + 2];
      c[0][tl] = wmma_bf16(a[0], bh, c[0][tl]);
      c[0][tl] = wmma_bf16(a[0], bl, c[0][tl]);
      c[1][tl] = wmma_bf16(a[1], bh, c[1][tl]);
      c[1][tl] = wmma_bf16(a[1], bl, c[1][tl]);
    }

    wait_async();       // my async writes to the next buffer have landed
    __syncthreads();    // everyone done reading cur + staging next
  }

  // epilogue: scatter C tiles to tail slice, deg from ones-tile column 0
  float* tailS = tail0 + (size_t)ksplit * GN * GD;
  float* degS  = deg0 + (size_t)ksplit * GN;
#pragma unroll
  for (int rt = 0; rt < 2; ++rt) {
    int rbase = rowBase + rt * 16 + ((lane < 16) ? 0 : 8);
#pragma unroll
    for (int tl = 0; tl < 8; ++tl) {
      int col = tl * 16 + lrow;
#pragma unroll
      for (int p = 0; p < 8; ++p)
        tailS[(size_t)(rbase + p) * GD + col] = c[rt][tl][p];
    }
    if (lrow == 0) {
#pragma unroll
      for (int p = 0; p < 8; ++p) degS[rbase + p] = cdeg[rt][p];
    }
  }
}

// ---------------------------------------------------------------------------
// Kernel 2: updated = tanh(f@Wo^T + bo - bd - (concat(f*deg, tail)/nn)@Wd^T)
// as one [N,384] x [384,256] GEMM; per-row scales folded into A; W'' bf16
// hi/lo planes precomputed -> B fragments are pure b128 loads (no VALU split).
// Grid: 256 blocks x 4 waves; wave = one 16-row tile, all 256 H columns.
// ---------------------------------------------------------------------------
__global__ __launch_bounds__(128, 1)
void fuse_kernel(const float* __restrict__ feat,
                 const unsigned short* __restrict__ Wbh,
                 const unsigned short* __restrict__ Wbl,
                 const float* __restrict__ bd, const float* __restrict__ bo,
                 const float* __restrict__ tail0, const float* __restrict__ deg0,
                 float* __restrict__ out) {
  const float* tail1 = tail0 + (size_t)GN * GD;
  const float* deg1  = deg0 + GN;
  const int lane  = threadIdx.x & 31;
  const int lrow  = lane & 15;
  const int r0    = (blockIdx.x * 4 + (threadIdx.x >> 5)) * 16;
  const int myrow = r0 + lrow;
  const int koffA  = (lane < 16) ? 0 : 8;
  const int khalfB = (lane < 16) ? 0 : 8;

  float dsum = deg0[myrow] + deg1[myrow];
  float nn   = fmaxf(dsum, 1.0f);
  float sh   = -dsum / nn;       // head scale (sign folds the subtraction)
  float st   = -1.0f / nn;       // tail scale

  v8f c[16];
#pragma unroll
  for (int t = 0; t < 16; ++t) c[t] = v8fzero();

  for (int kt = 0; kt < 12; ++kt) {
    const int k0 = kt * 32;
    float av[16];
    if (k0 < 128) {                               // head: f * (-deg/nn)
      load16gap(feat + (size_t)myrow * GD + k0 + koffA, av);
#pragma unroll
      for (int j = 0; j < 16; ++j) av[j] *= sh;
    } else if (k0 < 256) {                        // tail: (t0+t1) * (-1/nn)
      float t0v[16], t1v[16];
      load16gap(tail0 + (size_t)myrow * GD + (k0 - 128) + koffA, t0v);
      load16gap(tail1 + (size_t)myrow * GD + (k0 - 128) + koffA, t1v);
#pragma unroll
      for (int j = 0; j < 16; ++j) av[j] = (t0v[j] + t1v[j]) * st;
    } else {                                      // self term: f @ Wo^T
      load16gap(feat + (size_t)myrow * GD + (k0 - 256) + koffA, av);
    }
    BFrag ah, al;
    splitpack(av, ah, al);

    const int kB = k0 + khalfB;
#pragma unroll
    for (int ht = 0; ht < 16; ++ht) {
      const int h = ht * 16 + lrow;
      const uint4* wh = (const uint4*)(Wbh + (size_t)h * 384 + kB);
      const uint4* wl = (const uint4*)(Wbl + (size_t)h * 384 + kB);
      BFrag bh, bl;
      bh.q[0] = wh[0]; bh.q[1] = wh[2];
      bl.q[0] = wl[0]; bl.q[1] = wl[2];
      c[ht] = wmma_bf16(ah, bh, c[ht]);
      c[ht] = wmma_bf16(ah, bl, c[ht]);
      c[ht] = wmma_bf16(al, bh, c[ht]);
    }
  }

  const size_t off2 = (size_t)GN * GH;
#pragma unroll
  for (int ht = 0; ht < 16; ++ht) {
    const int h = ht * 16 + lrow;
    const float bias = bo[h] - bd[h];
    const int rb = r0 + ((lane < 16) ? 0 : 8);
#pragma unroll
    for (int p = 0; p < 8; ++p) {
      float u = tanhf(c[ht][p] + bias);
      size_t o = (size_t)(rb + p) * GH + h;
      out[o] = u;
      out[off2 + o] = u;
    }
  }
}

// ---------------------------------------------------------------------------
extern "C" void kernel_launch(void* const* d_in, const int* in_sizes, int n_in,
                              void* d_out, int out_size, void* d_ws, size_t ws_size,
                              hipStream_t stream) {
  const float* features = (const float*)d_in[0];
  const int*   adj      = (const int*)d_in[1];
  const float* Wd       = (const float*)d_in[2];
  const float* bd       = (const float*)d_in[3];
  const float* Wo       = (const float*)d_in[4];
  const float* bo       = (const float*)d_in[5];
  float* out = (float*)d_out;

  char* ws = (char*)d_ws;
  unsigned short* fTh = (unsigned short*)ws;                 // 4 MB
  unsigned short* fTl = fTh + (size_t)GD * GN;               // 4 MB
  float* tail0 = (float*)(ws + (size_t)8 * 1024 * 1024);     // 2 x 8 MB
  float* deg0  = tail0 + (size_t)2 * GN * GD;                // 2 x 64 KB
  unsigned short* Wbh = (unsigned short*)(deg0 + 2 * GN);    // 192 KB
  unsigned short* Wbl = Wbh + (size_t)GH * 384;              // 192 KB

  prep_kernel<<<(GN * GD) / 256, 256, 0, stream>>>(features, fTh, fTl);
  prepw_kernel<<<(GH * 384) / 256, 256, 0, stream>>>(Wd, Wo, Wbh, Wbl);
  gemm_adj_kernel<<<256, 128, 0, stream>>>(adj, fTh, fTl, tail0, deg0);
  fuse_kernel<<<256, 128, 0, stream>>>(features, Wbh, Wbl, bd, bo, tail0, deg0, out);
}
